// EdgeNet_39479339385306
// MI455X (gfx1250) — matrix operand order repaired and checked by reading, not compiled, not run on hardware
//
#include <hip/hip_runtime.h>
#include <hip/hip_bf16.h>

// ---------------------------------------------------------------------------
// EdgeNet (EdgeConv GNN) for MI455X / gfx1250, wave32.
//
// Algebraic rewrite: msg = sigmoid([xi | xj-xi] @ Wc.T + b)
//                        = sigmoid(P[dst] + Q[src]),
//   P = feats @ (W1-W2).T + b_c,  Q = feats @ W2.T   (node-level GEMMs)
// so the 19.7 GFLOP edge MLP becomes a 0.6 GFLOP/iter node GEMM (WMMA f16)
// plus a cheap gather/sigmoid/atomic-add edge phase that lives in L2.
// ---------------------------------------------------------------------------

typedef __attribute__((ext_vector_type(16))) _Float16 v16h;
typedef __attribute__((ext_vector_type(8)))  float    v8f;

#define D_IN  16
#define D_H   32
#define D_CAT 48
#define KPAD  64   // padded K (and feature-row stride in halfs)

// ---------------- input net: H = tanh(x @ W_in.T + b_in); F = [H | x | 0pad]
__global__ __launch_bounds__(256) void k_input(const float* __restrict__ x,
                                               const float* __restrict__ W_in,
                                               const float* __restrict__ b_in,
                                               _Float16* __restrict__ F, int N) {
  int t = blockIdx.x * blockDim.x + threadIdx.x;
  int n = t >> 5, c = t & 31;
  if (n >= N) return;
  const float* xr = x + (size_t)n * D_IN;
  const float* w  = W_in + c * D_IN;
  float s = b_in[c];
#pragma unroll
  for (int k = 0; k < D_IN; ++k) s += xr[k] * w[k];
  _Float16* fr = F + (size_t)n * KPAD;
  fr[c] = (_Float16)tanhf(s);
  if (c < D_IN) {                      // x part + zero padding written once
    fr[D_H + c]   = (_Float16)xr[c];
    fr[D_CAT + c] = (_Float16)0.f;
  }
}

// ---------------- build WT[j][k] (f16, 64x64): j<32 -> (W1-W2).T, j>=32 -> W2.T
__global__ void k_prep(const float* __restrict__ W_c, _Float16* __restrict__ WT) {
  int idx = blockIdx.x * blockDim.x + threadIdx.x;
  if (idx >= KPAD * KPAD) return;
  int j = idx >> 6, k = idx & 63;
  float v = 0.f;
  if (k < D_CAT) {
    if (j < D_H) v = W_c[j * 96 + k] - W_c[j * 96 + D_CAT + k];
    else         v = W_c[(j - D_H) * 96 + D_CAT + k];
  }
  WT[idx] = (_Float16)v;
}

// ---------------- PQ[n][0:64] = F[n][0:64] @ WT.T  via v_wmma_f32_16x16x32_f16
// one wave => 16 rows x 64 cols (4 col-tiles x 2 k-steps = 8 WMMAs)
__global__ __launch_bounds__(256) void k_pq(const _Float16* __restrict__ F,
                                            const _Float16* __restrict__ WT,
                                            const float* __restrict__ b_c,
                                            float* __restrict__ PQ, int N) {
  const int wave  = threadIdx.x >> 5;
  const int lane  = threadIdx.x & 31;
  const int tile  = blockIdx.x * 8 + wave;
  const int row0  = tile * 16;
  if (row0 >= N) return;                       // wave-uniform: EXEC stays all-1s
  const int  lh = lane & 15;
  const bool hi = lane >= 16;

  // A fragment (ISA 7.12.2, 16-bit A 16x32): lanes 0-15 K in {0..7,16..23},
  // lanes 16-31 K in {8..15,24..31}; two 16B loads per k-step.
  v16h a[2];
  {
    const _Float16* rowp = F + (size_t)(row0 + lh) * KPAD + (hi ? 8 : 0);
#pragma unroll
    for (int ks = 0; ks < 2; ++ks) {
      const _Float16* p = rowp + ks * 32;
      v16h av;
      ((float4*)&av)[0] = *(const float4*)(p);        // 8 halfs: K+0..7
      ((float4*)&av)[1] = *(const float4*)(p + 16);   // 8 halfs: K+16..23
      a[ks] = av;
    }
  }

  // B fragment (dense 16-bit B 32x16): lanes 0-15 hold K in {0..7,16..23} of column N=lane,
  // lanes 16-31 hold K in {8..15,24..31}. WT is stored [col][k] so K runs are contiguous.
  v16h b[4][2];
  {
    const int kb = hi ? 8 : 0;
#pragma unroll
    for (int ct = 0; ct < 4; ++ct) {
      const _Float16* colp = WT + (size_t)(ct * 16 + lh) * KPAD + kb;
#pragma unroll
      for (int ks = 0; ks < 2; ++ks) {
        const _Float16* p = colp + ks * 32;
        v16h bv;
        ((float4*)&bv)[0] = *(const float4*)(p);      // K+0..7
        ((float4*)&bv)[1] = *(const float4*)(p + 16); // K+16..23
        b[ct][ks] = bv;
      }
    }
  }

  v8f c0 = {}, c1 = {}, c2 = {}, c3 = {};
#pragma unroll
  for (int ks = 0; ks < 2; ++ks) {
    c0 = __builtin_amdgcn_wmma_f32_16x16x32_f16(false, a[ks], false, b[0][ks], (short)0, c0, false, false);
    c1 = __builtin_amdgcn_wmma_f32_16x16x32_f16(false, a[ks], false, b[1][ks], (short)0, c1, false, false);
    c2 = __builtin_amdgcn_wmma_f32_16x16x32_f16(false, a[ks], false, b[2][ks], (short)0, c2, false, false);
    c3 = __builtin_amdgcn_wmma_f32_16x16x32_f16(false, a[ks], false, b[3][ks], (short)0, c3, false, false);
  }

  // Epilogue: D layout (f32 C/D 16x16): lane<16 -> M=r, lane>=16 -> M=r+8, N=lane%16.
  // Fold b_c into the P half (cols 0..31).
  v8f acc[4] = {c0, c1, c2, c3};
#pragma unroll
  for (int ct = 0; ct < 4; ++ct) {
    const int col  = ct * 16 + lh;
    const float bias = (col < D_H) ? b_c[col] : 0.f;
#pragma unroll
    for (int r = 0; r < 8; ++r) {
      const int m = r + (hi ? 8 : 0);
      PQ[(size_t)(row0 + m) * KPAD + col] = acc[ct][r] + bias;
    }
  }
}

// ---------------- edge phase: Hnew[dst] += sigmoid(P[dst] + Q[src])
// 8 lanes per edge, float4 channel groups, fp32 global atomics (L2-resident).
__global__ __launch_bounds__(256) void k_edge(const int* __restrict__ ei,
                                              const float* __restrict__ PQ,
                                              float* __restrict__ Hnew, int E) {
  long long t = (long long)blockIdx.x * blockDim.x + threadIdx.x;
  if (t >= (long long)E * 8) return;
  const int e  = (int)(t >> 3);
  const int cg = ((int)t & 7) * 4;
  __builtin_prefetch(ei + e + 8192, 0, 0);          // global_prefetch_b8 on index stream
  const int s = ei[e];                              // src (x_j)
  const int d = ei[E + e];                          // dst (x_i / aggregation index)
  const float4 p = *(const float4*)(PQ + (size_t)d * KPAD + cg);
  const float4 q = *(const float4*)(PQ + (size_t)s * KPAD + D_H + cg);
  float4 h;
  h.x = 1.f / (1.f + __expf(-(p.x + q.x)));
  h.y = 1.f / (1.f + __expf(-(p.y + q.y)));
  h.z = 1.f / (1.f + __expf(-(p.z + q.z)));
  h.w = 1.f / (1.f + __expf(-(p.w + q.w)));
  float* o = Hnew + (size_t)d * D_H + cg;
  atomicAdd(o + 0, h.x);
  atomicAdd(o + 1, h.y);
  atomicAdd(o + 2, h.z);
  atomicAdd(o + 3, h.w);
}

// ---------------- refresh f16 feats H-part from Hnew (x-part persists)
__global__ __launch_bounds__(256) void k_feats(const float* __restrict__ Hnew,
                                               _Float16* __restrict__ F, int N) {
  int t = blockIdx.x * blockDim.x + threadIdx.x;
  int n = t >> 5, c = t & 31;
  if (n >= N) return;
  F[(size_t)n * KPAD + c] = (_Float16)Hnew[(size_t)n * D_H + c];
}

// ---------------- per-channel sums of H over nodes
__global__ __launch_bounds__(256) void k_reduce(const float* __restrict__ Hnew,
                                                float* __restrict__ sums, int N) {
  __shared__ float sh[256];
  const int c = threadIdx.x & 31;
  const int r = threadIdx.x >> 5;
  float acc = 0.f;
  for (int n = blockIdx.x * 8 + r; n < N; n += gridDim.x * 8)
    acc += Hnew[(size_t)n * D_H + c];
  sh[threadIdx.x] = acc;
  __syncthreads();
  if (threadIdx.x < 32) {
    float s = sh[threadIdx.x];
#pragma unroll
    for (int k = 1; k < 8; ++k) s += sh[k * 32 + threadIdx.x];
    atomicAdd(&sums[c], s);
  }
}

// ---------------- out = sigmoid(mean(H) @ W_out.T + b_out)
__global__ void k_final(const float* __restrict__ sums,
                        const float* __restrict__ W_out,
                        const float* __restrict__ b_out,
                        float* __restrict__ out, int N) {
  const int lane = threadIdx.x;           // 32 threads = one wave
  float v = (sums[lane] / (float)N) * W_out[lane];
#pragma unroll
  for (int off = 16; off > 0; off >>= 1) v += __shfl_xor(v, off, 32);
  if (lane == 0) out[0] = 1.f / (1.f + __expf(-(v + b_out[0])));
}

// ---------------------------------------------------------------------------
extern "C" void kernel_launch(void* const* d_in, const int* in_sizes, int n_in,
                              void* d_out, int out_size, void* d_ws, size_t ws_size,
                              hipStream_t stream) {
  const float* x     = (const float*)d_in[0];
  const int*   ei    = (const int*)  d_in[1];
  const float* W_in  = (const float*)d_in[2];
  const float* b_in  = (const float*)d_in[3];
  const float* W_c   = (const float*)d_in[4];
  const float* b_c   = (const float*)d_in[5];
  const float* W_out = (const float*)d_in[6];
  const float* b_out = (const float*)d_in[7];
  float* out = (float*)d_out;

  const int N = in_sizes[0] / D_IN;
  const int E = in_sizes[1] / 2;

  // workspace carve-out (256B aligned)
  char* ws = (char*)d_ws;
  size_t off = 0;
  auto carve = [&](size_t bytes) {
    char* p = ws + off;
    off = (off + bytes + 255) & ~(size_t)255;
    return p;
  };
  _Float16* F    = (_Float16*)carve((size_t)N * KPAD * sizeof(_Float16)); // 12.8 MB
  _Float16* WT   = (_Float16*)carve((size_t)KPAD * KPAD * sizeof(_Float16));
  float*    PQ   = (float*)   carve((size_t)N * KPAD * sizeof(float));    // 25.6 MB
  float*    Hnew = (float*)   carve((size_t)N * D_H * sizeof(float));     // 12.8 MB
  float*    sums = (float*)   carve(D_H * sizeof(float));
  (void)ws_size;

  const int nThreads = 256;
  const int gInput  = (N * 32 + nThreads - 1) / nThreads;
  const int gPq     = ((N + 15) / 16 + 7) / 8;
  const long long eWork = (long long)E * 8;
  const int gEdge   = (int)((eWork + nThreads - 1) / nThreads);
  const int gFeats  = gInput;

  k_input<<<gInput, nThreads, 0, stream>>>(x, W_in, b_in, F, N);
  k_prep<<<(KPAD * KPAD + nThreads - 1) / nThreads, nThreads, 0, stream>>>(W_c, WT);

  for (int it = 0; it < 2; ++it) {
    k_pq<<<gPq, nThreads, 0, stream>>>(F, WT, b_c, PQ, N);
    hipMemsetAsync(Hnew, 0, (size_t)N * D_H * sizeof(float), stream);
    k_edge<<<gEdge, nThreads, 0, stream>>>(ei, PQ, Hnew, E);
    if (it == 0) k_feats<<<gFeats, nThreads, 0, stream>>>(Hnew, F, N);
  }

  hipMemsetAsync(sums, 0, D_H * sizeof(float), stream);
  k_reduce<<<1024, nThreads, 0, stream>>>(Hnew, sums, N);
  k_final<<<1, 32, 0, stream>>>(sums, W_out, b_out, out, N);
  (void)out_size; (void)n_in;
}
